// LSTMModel_16346645529042
// MI455X (gfx1250) — compile-verified, matrix operand order checked by reading, not run on hardware
//
#include <hip/hip_runtime.h>
#include <hip/hip_bf16.h>

// ---------------------------------------------------------------------------
// 2-layer LSTM (H=256, B=512, T=512) + FC(6) for MI455X (gfx1250, wave32).
//
// Recurrence is batch-independent -> each workgroup owns a 16-row batch tile
// and loops over all T steps privately.  Per step:
//   gates[16,1024] = xg + h[16,256] @ Whh^T   via v_wmma_f32_16x16x32_f16
// A-operand (h tile, f16) staged in LDS (ds_load_b128); B-operand (permuted
// f16 weights, 512 KB) streamed from L2 each step (fully resident in 192 MB
// L2) as global_load_b128.  16 waves/block, each wave owns 16 hidden channels
// x 4 gates = 64 N-columns -> low register pressure, lane-local cell update.
// An opaque SGPR offset (volatile asm) inside the t-loop stops LICM from
// hoisting the time-invariant weight fragments (round-1 showed scratch
// spills) while preserving pointer provenance so loads stay GLOBAL, not FLAT
// (round-2 laundering via uintptr_t degraded them to flat_load, which ties
// up DScnt alongside the LDS A-fragment traffic).
// ---------------------------------------------------------------------------

#define Bsz   512
#define Tlen  512
#define Hdim  256
#define G4    1024   // 4*H
#define BT    16     // batch rows per workgroup
#define NTPW  4      // N-tiles per wave (4 gates x 16 channels)

typedef _Float16 v8h  __attribute__((ext_vector_type(8)));
typedef _Float16 v16h __attribute__((ext_vector_type(16)));
typedef float    v8f  __attribute__((ext_vector_type(8)));

static __device__ __forceinline__ v16h load_frag16(const _Float16* p) {
  // WMMA 16-bit A/B fragment: two contiguous 16-byte chunks per lane
  // (halves k0..k0+7 and k0+16..k0+23) -> ds_load_b128 / global_load_b128.
  v8h lo = *(const v8h*)(p);
  v8h hi = *(const v8h*)(p + 16);
  return __builtin_shufflevector(lo, hi, 0,1,2,3,4,5,6,7,8,9,10,11,12,13,14,15);
}

// Opaque zero in an SGPR: regenerated inside the t-loop (volatile asm cannot
// be hoisted), so weight addresses cannot be proven loop-invariant, but the
// base pointer keeps its global-address-space provenance.
static __device__ __forceinline__ int opaque_zero() {
  int z = 0;
  asm volatile("" : "+s"(z));
  return z;
}

static __device__ __forceinline__ float sigm(float x) {
  // fast path: v_exp_f32 + v_add + v_rcp_f32 (no IEEE div fixup sequence)
  return __builtin_amdgcn_rcpf(1.0f + __expf(-x));
}
static __device__ __forceinline__ float ftanh(float x) {
  float e = __expf(2.0f * x);            // saturates cleanly at +/-inf
  return 1.0f - 2.0f * __builtin_amdgcn_rcpf(e + 1.0f);
}

// ---------------------------------------------------------------------------
// Prep: permute weight rows and convert to f16; fold biases.
// p in [0,1024): wv=p>>6, r=p&63, gate=r>>4, jj=r&15,
//                orig_row = gate*256 + wv*16 + jj.
// Wave wv then owns hidden channels [16wv,16wv+16) with all 4 gates adjacent.
// ---------------------------------------------------------------------------
__global__ void __launch_bounds__(256)
prep_weights(const float* __restrict__ Wih0, const float* __restrict__ Whh0,
             const float* __restrict__ bih0, const float* __restrict__ bhh0,
             const float* __restrict__ Wih1, const float* __restrict__ Whh1,
             const float* __restrict__ bih1, const float* __restrict__ bhh1,
             _Float16* __restrict__ w0h, _Float16* __restrict__ w1i,
             _Float16* __restrict__ w1h, float* __restrict__ wih0p,
             float* __restrict__ b0p, float* __restrict__ b1p) {
  int p = blockIdx.x;                 // 0..1023 permuted row
  int k = threadIdx.x;                // 0..255 K index (unpermuted)
  int wv = p >> 6, r = p & 63, gate = r >> 4, jj = r & 15;
  int o = gate * 256 + wv * 16 + jj;  // original row

  w0h[(size_t)p * 256 + k] = (_Float16)Whh0[(size_t)o * 256 + k];
  w1i[(size_t)p * 256 + k] = (_Float16)Wih1[(size_t)o * 256 + k];
  w1h[(size_t)p * 256 + k] = (_Float16)Whh1[(size_t)o * 256 + k];
  if (k < 3)  wih0p[p * 3 + k] = Wih0[o * 3 + k];   // D=3 input proj, keep f32
  if (k == 0) { b0p[p] = bih0[o] + bhh0[o]; b1p[p] = bih1[o] + bhh1[o]; }
}

// ---------------------------------------------------------------------------
// Layer 0: fused input projection (D=3, VALU) + recurrent WMMA GEMM.
// grid = 32 workgroups (512/16 batch rows), 512 threads = 16 waves.
// Writes ys0[B,T,H] in f16 (layer-1 input).
// ---------------------------------------------------------------------------
__global__ void __launch_bounds__(512)
lstm_layer0(const float* __restrict__ x, const _Float16* __restrict__ w0h,
            const float* __restrict__ wih0p, const float* __restrict__ b0p,
            _Float16* __restrict__ ys0) {
  __shared__ __align__(16) _Float16 hsh[BT * Hdim];   // 8 KB h tile (f16)

  const int tid  = threadIdx.x;
  const int l    = tid & 31;
  const int wv   = tid >> 5;          // wave id 0..15
  const int l16  = l & 15;
  const int hsel = l >> 4;            // lane half -> K-chunk select
  const int bbase = blockIdx.x * BT;

  for (int i = tid; i < BT * Hdim; i += 512) hsh[i] = (_Float16)0.0f;

  // Per-lane constants: 4 N-tiles (= gates) -> permuted row, bias, 3 weights.
  int   pcol[NTPW];
  float bv[NTPW], wiv[NTPW][3];
#pragma unroll
  for (int nt = 0; nt < NTPW; ++nt) {
    int p = wv * 64 + nt * 16 + l16;
    pcol[nt]   = p;
    bv[nt]     = b0p[p];
    wiv[nt][0] = wih0p[p * 3 + 0];
    wiv[nt][1] = wih0p[p * 3 + 1];
    wiv[nt][2] = wih0p[p * 3 + 2];
  }

  float cst[8];
#pragma unroll
  for (int r = 0; r < 8; ++r) cst[r] = 0.0f;

  const int j = wv * 16 + l16;        // this lane's original hidden channel
  // x rows for this lane: batch rows m = r + 8*hsel; stride Tlen*3 floats.
  const float* xrow0 = x + (size_t)(bbase + 8 * hsel) * Tlen * 3;
  // ys0 rows: stride Tlen*Hdim halves.
  _Float16* yrow0 = ys0 + ((size_t)(bbase + 8 * hsel) * Tlen) * Hdim + j;

  __syncthreads();

  for (int t = 0; t < Tlen; ++t) {
    const _Float16* w0 = w0h + opaque_zero();   // keep B-frags inside t-loop

    // ---- accumulator init: bias + x @ Wih0^T (D=3, pure VALU) ----
    v8f acc[NTPW];
    const float* xp = xrow0 + t * 3;
#pragma unroll
    for (int r = 0; r < 8; ++r) {
      float x0 = xp[r * (Tlen * 3) + 0];
      float x1 = xp[r * (Tlen * 3) + 1];
      float x2 = xp[r * (Tlen * 3) + 2];
#pragma unroll
      for (int nt = 0; nt < NTPW; ++nt)
        acc[nt][r] = bv[nt] + x0 * wiv[nt][0] + x1 * wiv[nt][1] + x2 * wiv[nt][2];
    }
    if (t + 1 < Tlen) __builtin_prefetch(xp + 3, 0, 3);

    // ---- gates += h @ Whh^T : K=256 in 8 WMMA steps of 32 ----
#pragma unroll 2
    for (int kk = 0; kk < Hdim; kk += 32) {
      v16h A = load_frag16(&hsh[l16 * Hdim + kk + hsel * 8]);       // ds_load
#pragma unroll
      for (int nt = 0; nt < NTPW; ++nt) {
        v16h Bf = load_frag16(&w0[(size_t)pcol[nt] * Hdim + kk + hsel * 8]);
        acc[nt] = __builtin_amdgcn_wmma_f32_16x16x32_f16(
            false, A, false, Bf, (short)0, acc[nt], false, false);
      }
    }
    __syncthreads();   // all waves finished reading old h tile

    // ---- LSTM cell update (lane-local thanks to gate permutation) ----
#pragma unroll
    for (int r = 0; r < 8; ++r) {
      float iv = sigm(acc[0][r]);
      float fv = sigm(acc[1][r]);
      float gv = ftanh(acc[2][r]);
      float ov = sigm(acc[3][r]);
      float c  = fv * cst[r] + iv * gv;
      cst[r] = c;
      float h  = ov * ftanh(c);
      int m = r + 8 * hsel;
      _Float16 hf = (_Float16)h;
      hsh[m * Hdim + j] = hf;
      yrow0[(size_t)r * (Tlen * Hdim) + (size_t)t * Hdim] = hf;
    }
    __syncthreads();   // new h tile visible before next step's reads
  }
}

// ---------------------------------------------------------------------------
// Layer 1: input projection fused into the K-loop (K=256 from global ys0,
// then K=256 from LDS h).  Stores last-step h (f32) for the FC epilogue.
// ---------------------------------------------------------------------------
__global__ void __launch_bounds__(512)
lstm_layer1(const _Float16* __restrict__ ys0, const _Float16* __restrict__ w1i,
            const _Float16* __restrict__ w1h, const float* __restrict__ b1p,
            float* __restrict__ hlast) {
  __shared__ __align__(16) _Float16 hsh[BT * Hdim];

  const int tid  = threadIdx.x;
  const int l    = tid & 31;
  const int wv   = tid >> 5;
  const int l16  = l & 15;
  const int hsel = l >> 4;
  const int bbase = blockIdx.x * BT;

  for (int i = tid; i < BT * Hdim; i += 512) hsh[i] = (_Float16)0.0f;

  int   pcol[NTPW];
  float bv[NTPW];
#pragma unroll
  for (int nt = 0; nt < NTPW; ++nt) {
    int p = wv * 64 + nt * 16 + l16;
    pcol[nt] = p;
    bv[nt]   = b1p[p];
  }

  float cst[8];
#pragma unroll
  for (int r = 0; r < 8; ++r) cst[r] = 0.0f;

  const int j = wv * 16 + l16;
  // A-fragment source row in ys0 for this lane (row = batch m = l16).
  const _Float16* arow = ys0 + ((size_t)(bbase + l16) * Tlen) * Hdim + hsel * 8;

  __syncthreads();

  for (int t = 0; t < Tlen; ++t) {
    int zk = opaque_zero();
    const _Float16* wi = w1i + zk;
    const _Float16* wh = w1h + zk;

    v8f acc[NTPW];
#pragma unroll
    for (int nt = 0; nt < NTPW; ++nt) {
      v8f z = {bv[nt], bv[nt], bv[nt], bv[nt], bv[nt], bv[nt], bv[nt], bv[nt]};
      acc[nt] = z;
    }

    // ---- phase 1: gates += ys0_t @ Wih1^T (A from global / L2) ----
    const _Float16* abase = arow + (size_t)t * Hdim;
    if (t + 1 < Tlen) __builtin_prefetch(abase + Hdim, 0, 3);
#pragma unroll 2
    for (int kk = 0; kk < Hdim; kk += 32) {
      v16h A = load_frag16(abase + kk);                             // global
#pragma unroll
      for (int nt = 0; nt < NTPW; ++nt) {
        v16h Bf = load_frag16(&wi[(size_t)pcol[nt] * Hdim + kk + hsel * 8]);
        acc[nt] = __builtin_amdgcn_wmma_f32_16x16x32_f16(
            false, A, false, Bf, (short)0, acc[nt], false, false);
      }
    }

    // ---- phase 2: gates += h @ Whh1^T (A from LDS) ----
#pragma unroll 2
    for (int kk = 0; kk < Hdim; kk += 32) {
      v16h A = load_frag16(&hsh[l16 * Hdim + kk + hsel * 8]);       // ds_load
#pragma unroll
      for (int nt = 0; nt < NTPW; ++nt) {
        v16h Bf = load_frag16(&wh[(size_t)pcol[nt] * Hdim + kk + hsel * 8]);
        acc[nt] = __builtin_amdgcn_wmma_f32_16x16x32_f16(
            false, A, false, Bf, (short)0, acc[nt], false, false);
      }
    }
    __syncthreads();

#pragma unroll
    for (int r = 0; r < 8; ++r) {
      float iv = sigm(acc[0][r]);
      float fv = sigm(acc[1][r]);
      float gv = ftanh(acc[2][r]);
      float ov = sigm(acc[3][r]);
      float c  = fv * cst[r] + iv * gv;
      cst[r] = c;
      float h  = ov * ftanh(c);
      int m = r + 8 * hsel;
      hsh[m * Hdim + j] = (_Float16)h;
      if (t == Tlen - 1)
        hlast[(size_t)(bbase + m) * Hdim + j] = h;   // original channel order
    }
    __syncthreads();
  }
}

// ---------------------------------------------------------------------------
// FC epilogue: out[b,j] = hlast[b,:] . Wfc[j,:] + bfc[j]   (512 x 6, f32)
// ---------------------------------------------------------------------------
__global__ void __launch_bounds__(256)
fc_kernel(const float* __restrict__ hlast, const float* __restrict__ Wfc,
          const float* __restrict__ bfc, float* __restrict__ out) {
  int idx = blockIdx.x * blockDim.x + threadIdx.x;
  if (idx >= Bsz * 6) return;
  int b = idx / 6, j = idx % 6;
  float s = bfc[j];
  const float* hp = &hlast[(size_t)b * Hdim];
  const float* wp = &Wfc[(size_t)j * Hdim];
#pragma unroll 8
  for (int k = 0; k < Hdim; ++k) s += hp[k] * wp[k];
  out[idx] = s;
}

// ---------------------------------------------------------------------------
extern "C" void kernel_launch(void* const* d_in, const int* in_sizes, int n_in,
                              void* d_out, int out_size, void* d_ws, size_t ws_size,
                              hipStream_t stream) {
  const float* x    = (const float*)d_in[0];
  const float* Wih0 = (const float*)d_in[1];
  const float* Whh0 = (const float*)d_in[2];
  const float* bih0 = (const float*)d_in[3];
  const float* bhh0 = (const float*)d_in[4];
  const float* Wih1 = (const float*)d_in[5];
  const float* Whh1 = (const float*)d_in[6];
  const float* bih1 = (const float*)d_in[7];
  const float* bhh1 = (const float*)d_in[8];
  const float* Wfc  = (const float*)d_in[9];
  const float* bfc  = (const float*)d_in[10];

  // Workspace layout (bytes)
  char* ws = (char*)d_ws;
  _Float16* w0h   = (_Float16*)(ws + 0);                 //  512 KB
  _Float16* w1i   = (_Float16*)(ws + 524288);            //  512 KB
  _Float16* w1h   = (_Float16*)(ws + 1048576);           //  512 KB
  float*    wih0p = (float*)   (ws + 1572864);           //   12 KB
  float*    b0p   = (float*)   (ws + 1585152);           //    4 KB
  float*    b1p   = (float*)   (ws + 1589248);           //    4 KB
  float*    hlast = (float*)   (ws + 1593344);           //  512 KB
  _Float16* ys0   = (_Float16*)(ws + 2117632);           //  128 MB [B,T,H] f16

  prep_weights<<<dim3(G4), dim3(256), 0, stream>>>(
      Wih0, Whh0, bih0, bhh0, Wih1, Whh1, bih1, bhh1,
      w0h, w1i, w1h, wih0p, b0p, b1p);

  lstm_layer0<<<dim3(Bsz / BT), dim3(512), 0, stream>>>(x, w0h, wih0p, b0p, ys0);

  lstm_layer1<<<dim3(Bsz / BT), dim3(512), 0, stream>>>(ys0, w1i, w1h, b1p, hlast);

  fc_kernel<<<dim3((Bsz * 6 + 255) / 256), dim3(256), 0, stream>>>(
      hlast, Wfc, bfc, (float*)d_out);
}
